// GraphNet_44014824849589
// MI455X (gfx1250) — matrix operand order repaired and checked by reading, not compiled, not run on hardware
//
#include <hip/hip_runtime.h>
#include <hip/hip_bf16.h>

// ---------------- problem constants (match reference module) ----------------
#define NN        100000
#define EE        1600000
#define NUM_DOCS  20000
#define D_RAW     300
#define D_IN      768
#define D_H       200
#define D_OUT     8
#define N_WORDS   (NN - NUM_DOCS)

typedef __attribute__((ext_vector_type(16))) __bf16 v16bf;
typedef __attribute__((ext_vector_type(8)))  float  v8f;

static __device__ __forceinline__ unsigned pack2(__bf16 a, __bf16 b) {
    unsigned short ua = __builtin_bit_cast(unsigned short, a);
    unsigned short ub = __builtin_bit_cast(unsigned short, b);
    return ((unsigned)ub << 16) | (unsigned)ua;
}

#define WMMA_BF16(A_, B_, C_) \
    __builtin_amdgcn_wmma_f32_16x16x32_bf16(false, (A_), false, (B_), \
                                            (short)0, (C_), false, false)

// ======================= WMMA GEMM: C = A(MxK) * B(KxNn) [+bias] ============
// Block tile 64(M) x 64(N), K-step 32. 128 threads = 4 waves; wave w owns
// row-subtile w (16 rows) and iterates the four 16-col subtiles, reusing its
// A fragment across 4 B tiles -> 12 WMMAs / wave / k-step.
//
// fp32 is split into bf16 hi/lo; product = hi*hi + hi*lo + lo*hi (3 WMMAs,
// ~fp32 accuracy, fp32 accumulate). LDS holds data in *fragment-major* order:
// [subtile][lane][element], so each lane loads its whole v16bf fragment as a
// contiguous 32-byte vector (2x ds_load_b128) instead of 16 scattered u16s.
//
// Layout inversions (per CDNA5 ISA 7.12.2, wave32):
//   A 16x32 : lane = (M&15) + 16*((K>>3)&1), elem = (K>>4)*8 + (K&7)
//   B 32x16 : lane = (N&15) + 16*((K>>3)&1), elem = (K>>4)*8 + (K&7)
//   C 16x16 : VGPR v -> M = v + 8*(lane>>4), N = lane & 15
__global__ __launch_bounds__(128)
void gemm_bf16x3(const float* __restrict__ A, int lda,
                 const float* __restrict__ B, int ldb,
                 float* __restrict__ C, int ldc,
                 const float* __restrict__ bias,
                 int M, int Nn, int K)
{
    __shared__ __align__(32) __bf16 sa_hi[4][32][16];
    __shared__ __align__(32) __bf16 sa_lo[4][32][16];
    __shared__ __align__(32) __bf16 sb_hi[4][32][16];
    __shared__ __align__(32) __bf16 sb_lo[4][32][16];

    const int tid  = threadIdx.x;
    const int lane = tid & 31;
    const int wave = tid >> 5;      // 0..3 : row subtile
    const int l    = lane & 15;
    const int h    = lane >> 4;

    const int m_base = blockIdx.y * 64;
    const int n_base = blockIdx.x * 64;

    v8f acc[4] = {};

    for (int k0 = 0; k0 < K; k0 += 32) {
        // ---- stage A 64x32: K-pairs, float2 global loads, packed b32 LDS ----
        #pragma unroll
        for (int i = 0; i < 8; ++i) {
            int p = tid + i * 128;          // pair index 0..1023
            int r = p >> 4;                 // row in tile 0..63
            int c = (p & 15) << 1;          // even K col 0..30
            float2 v = make_float2(0.0f, 0.0f);
            int gm = m_base + r, gk = k0 + c;
            if (gm < M && gk < K)           // K even -> pair fully valid
                v = *(const float2*)(A + (size_t)gm * lda + gk);
            __bf16 h0 = (__bf16)v.x; __bf16 l0 = (__bf16)(v.x - (float)h0);
            __bf16 h1 = (__bf16)v.y; __bf16 l1 = (__bf16)(v.y - (float)h1);
            int t  = r >> 4;
            int ln = (r & 15) + (((c >> 3) & 1) << 4);
            int e  = ((c >> 4) << 3) + (c & 7);          // even
            *(unsigned*)&sa_hi[t][ln][e] = pack2(h0, h1);
            *(unsigned*)&sa_lo[t][ln][e] = pack2(l0, l1);
        }
        // ---- stage B 32x64: K-row pairs, packed b32 LDS ----
        #pragma unroll
        for (int i = 0; i < 8; ++i) {
            int p  = tid + i * 128;         // pair index 0..1023
            int c  = p & 63;                // tile col 0..63
            int kr = (p >> 6) << 1;         // even K row 0..30
            float v0 = 0.0f, v1 = 0.0f;
            int gk = k0 + kr, gn = n_base + c;
            if (gk < K && gn < Nn) {        // K even -> pair fully valid
                v0 = B[(size_t)gk * ldb + gn];
                v1 = B[(size_t)(gk + 1) * ldb + gn];
            }
            __bf16 h0 = (__bf16)v0; __bf16 l0 = (__bf16)(v0 - (float)h0);
            __bf16 h1 = (__bf16)v1; __bf16 l1 = (__bf16)(v1 - (float)h1);
            int sub = c >> 4;
            int ln  = (c & 15) + (((kr >> 3) & 1) << 4);
            int e   = ((kr >> 4) << 3) + (kr & 7);       // even
            *(unsigned*)&sb_hi[sub][ln][e] = pack2(h0, h1);
            *(unsigned*)&sb_lo[sub][ln][e] = pack2(l0, l1);
        }
        __syncthreads();

        // ---- fragments: one contiguous v16bf per lane ----
        v16bf ah = *(const v16bf*)&sa_hi[wave][lane][0];
        v16bf al = *(const v16bf*)&sa_lo[wave][lane][0];
        #pragma unroll
        for (int sub = 0; sub < 4; ++sub) {
            v16bf bh = *(const v16bf*)&sb_hi[sub][lane][0];
            v16bf bl = *(const v16bf*)&sb_lo[sub][lane][0];
            acc[sub] = WMMA_BF16(ah, bh, acc[sub]);
            acc[sub] = WMMA_BF16(ah, bl, acc[sub]);
            acc[sub] = WMMA_BF16(al, bh, acc[sub]);
        }
        __syncthreads();
    }

    // ---- store 4 x (16x16) tiles ----
    #pragma unroll
    for (int sub = 0; sub < 4; ++sub) {
        #pragma unroll
        for (int vv = 0; vv < 8; ++vv) {
            int m = m_base + wave * 16 + vv + 8 * h;
            int n = n_base + sub * 16 + l;
            if (m < M && n < Nn) {
                float o = acc[sub][vv];
                if (bias) o += bias[n];
                C[(size_t)m * ldc + n] = o;
            }
        }
    }
}

// ============================ degree / norm =================================
__global__ void deg_init(float* __restrict__ deg) {
    int i = blockIdx.x * blockDim.x + threadIdx.x;
    if (i < NN) deg[i] = 1.0f;                       // self-loop weight
}

__global__ void deg_edges(const int* __restrict__ ei, const float* __restrict__ ew,
                          float* __restrict__ deg) {
    int e = blockIdx.x * blockDim.x + threadIdx.x;
    if (e < EE) {
        int c = ei[EE + e];                           // target node
        __hip_atomic_fetch_add(&deg[c], ew[e], __ATOMIC_RELAXED,
                               __HIP_MEMORY_SCOPE_AGENT);
    }
}

__global__ void dinv_kernel(const float* __restrict__ deg, float* __restrict__ dinv) {
    int i = blockIdx.x * blockDim.x + threadIdx.x;
    if (i < NN) {
        float d = deg[i];
        dinv[i] = (d > 0.0f) ? rsqrtf(d) : 0.0f;
    }
}

// ==================== aggregation: init (self loop + bias) ==================
__global__ void agg1_init(const float* __restrict__ h1, const float* __restrict__ dinv,
                          const float* __restrict__ b1, float* __restrict__ a1) {
    long long gid = (long long)blockIdx.x * blockDim.x + threadIdx.x;
    if (gid < (long long)NN * D_H) {
        int i = (int)(gid / D_H);
        int d = (int)(gid % D_H);
        float di = dinv[i];
        a1[gid] = h1[gid] * di * di + b1[d];
    }
}

// wave per edge, 200 dims strided across 32 lanes
__global__ void agg1_edges(const int* __restrict__ ei, const float* __restrict__ ew,
                           const float* __restrict__ dinv,
                           const float* __restrict__ h1, float* __restrict__ a1) {
    long long gid = (long long)blockIdx.x * blockDim.x + threadIdx.x;
    int e = (int)(gid >> 5);
    int lane = (int)(gid & 31);
    if (e >= EE) return;
    int r = ei[e];
    int c = ei[EE + e];
    float norm = dinv[r] * ew[e] * dinv[c];
    const float* src = h1 + (size_t)r * D_H;
    float* dst       = a1 + (size_t)c * D_H;
    for (int d = lane; d < D_H; d += 32) {
        __hip_atomic_fetch_add(dst + d, src[d] * norm, __ATOMIC_RELAXED,
                               __HIP_MEMORY_SCOPE_AGENT);
    }
}

__global__ void relu_kernel(float* __restrict__ x, long long n) {
    long long gid = (long long)blockIdx.x * blockDim.x + threadIdx.x;
    if (gid < n) x[gid] = fmaxf(x[gid], 0.0f);
}

// ======================= small GEMM: h2 = a1 @ w2 (200->8) ==================
__global__ __launch_bounds__(256)
void linear_small(const float* __restrict__ A, const float* __restrict__ W,
                  float* __restrict__ O) {
    __shared__ float sw[D_H * D_OUT];
    for (int i = threadIdx.x; i < D_H * D_OUT; i += 256) sw[i] = W[i];
    __syncthreads();
    int i = blockIdx.x * 256 + threadIdx.x;
    if (i >= NN) return;
    const float* a = A + (size_t)i * D_H;
    float acc[D_OUT] = {};
    for (int k = 0; k < D_H; ++k) {
        float av = a[k];
        #pragma unroll
        for (int d = 0; d < D_OUT; ++d) acc[d] += av * sw[k * D_OUT + d];
    }
    #pragma unroll
    for (int d = 0; d < D_OUT; ++d) O[(size_t)i * D_OUT + d] = acc[d];
}

__global__ void agg2_init(const float* __restrict__ h2, const float* __restrict__ dinv,
                          const float* __restrict__ b2, float* __restrict__ out) {
    int gid = blockIdx.x * blockDim.x + threadIdx.x;
    if (gid < NN * D_OUT) {
        int i = gid >> 3;
        int d = gid & 7;
        float di = dinv[i];
        out[gid] = h2[gid] * di * di + b2[d];
    }
}

// thread per (edge, dim)
__global__ void agg2_edges(const int* __restrict__ ei, const float* __restrict__ ew,
                           const float* __restrict__ dinv,
                           const float* __restrict__ h2, float* __restrict__ out) {
    long long gid = (long long)blockIdx.x * blockDim.x + threadIdx.x;
    int e = (int)(gid >> 3);
    int d = (int)(gid & 7);
    if (e >= EE) return;
    int r = ei[e];
    int c = ei[EE + e];
    float norm = dinv[r] * ew[e] * dinv[c];
    __hip_atomic_fetch_add(out + (size_t)c * D_OUT + d,
                           h2[(size_t)r * D_OUT + d] * norm,
                           __ATOMIC_RELAXED, __HIP_MEMORY_SCOPE_AGENT);
}

// ================================ launch ====================================
extern "C" void kernel_launch(void* const* d_in, const int* in_sizes, int n_in,
                              void* d_out, int out_size, void* d_ws, size_t ws_size,
                              hipStream_t stream) {
    const float* features = (const float*)d_in[0];   // N x 768
    const int*   ei       = (const int*)d_in[1];     // 2 x E
    const float* ew       = (const float*)d_in[2];   // E
    const float* lin_w    = (const float*)d_in[4];   // 300 x 768
    const float* lin_b    = (const float*)d_in[5];   // 768
    const float* w1       = (const float*)d_in[6];   // 768 x 200
    const float* b1       = (const float*)d_in[7];   // 200
    const float* w2       = (const float*)d_in[8];   // 200 x 8
    const float* b2       = (const float*)d_in[9];   // 8
    float* out = (float*)d_out;                      // N x 8

    // workspace layout (floats)
    float* x0   = (float*)d_ws;                      // N*768
    float* h1   = x0  + (size_t)NN * D_IN;           // N*200
    float* a1   = h1  + (size_t)NN * D_H;            // N*200
    float* h2   = a1  + (size_t)NN * D_H;            // N*8
    float* deg  = h2  + (size_t)NN * D_OUT;          // N
    float* dinv = deg + NN;                          // N

    const int T = 256;

    // --- x0: doc rows are a straight copy; word rows via WMMA GEMM ---
    hipMemcpyAsync(x0, features, (size_t)NUM_DOCS * D_IN * sizeof(float),
                   hipMemcpyDeviceToDevice, stream);
    {
        dim3 g((D_IN + 63) / 64, (N_WORDS + 63) / 64);          // 12 x 1250
        gemm_bf16x3<<<g, 128, 0, stream>>>(
            features + (size_t)NUM_DOCS * D_IN, D_IN,   // A: word raw feats (K=300 of 768-stride rows)
            lin_w, D_IN,                                // B: 300 x 768
            x0 + (size_t)NUM_DOCS * D_IN, D_IN,         // C
            lin_b, N_WORDS, D_IN, D_RAW);
    }

    // --- symmetric normalization coefficients ---
    deg_init<<<(NN + T - 1) / T, T, 0, stream>>>(deg);
    deg_edges<<<(EE + T - 1) / T, T, 0, stream>>>(ei, ew, deg);
    dinv_kernel<<<(NN + T - 1) / T, T, 0, stream>>>(deg, dinv);

    // --- layer 1: h1 = x0 @ w1 (WMMA), aggregate, +b1, relu ---
    {
        dim3 g((D_H + 63) / 64, (NN + 63) / 64);                // 4 x 1563
        gemm_bf16x3<<<g, 128, 0, stream>>>(x0, D_IN, w1, D_H, h1, D_H,
                                           nullptr, NN, D_H, D_IN);
    }
    {
        long long n = (long long)NN * D_H;
        agg1_init<<<(unsigned)((n + T - 1) / T), T, 0, stream>>>(h1, dinv, b1, a1);
        long long waves = (long long)EE * 32;
        agg1_edges<<<(unsigned)((waves + T - 1) / T), T, 0, stream>>>(ei, ew, dinv, h1, a1);
        relu_kernel<<<(unsigned)((n + T - 1) / T), T, 0, stream>>>(a1, n);
    }

    // --- layer 2: h2 = a1 @ w2 (tiny), aggregate into d_out, +b2 ---
    linear_small<<<(NN + T - 1) / T, T, 0, stream>>>(a1, w2, h2);
    agg2_init<<<(NN * D_OUT + T - 1) / T, T, 0, stream>>>(h2, dinv, b2, out);
    {
        long long thr = (long long)EE * D_OUT;
        agg2_edges<<<(unsigned)((thr + T - 1) / T), T, 0, stream>>>(ei, ew, dinv, h2, out);
    }
}
